// WindowAttentionGlobal_81690277970127
// MI455X (gfx1250) — matrix operand order repaired and checked by reading, not compiled, not run on hardware
//
#include <hip/hip_runtime.h>
#include <hip/hip_bf16.h>

// MI455X / gfx1250: all GEMM work on v_wmma_f32_16x16x32_f16 (f16 in, f32 acc).
// wave32, 256-thread (8-wave) workgroups. 2x2 register tiling per wave in the
// GEMMs; contiguous vector fragment loads matching the ISA VGPR layouts.
//
// Workspace (f16): Kws [b][h][343][32] | Vt [b][h][32][352] | Aws [43904][512]

typedef __attribute__((ext_vector_type(16))) _Float16 v16h;
typedef __attribute__((ext_vector_type(8)))  _Float16 v8h;
typedef __attribute__((ext_vector_type(8)))  float    v8f;

#define NHEADS 16
#define HD     32
#define NTOK   343
#define NPAD   352   // 22 * 16
#define VSTR   352   // Vt row stride (halves)
#define CDIM   512

__device__ __forceinline__ v16h cat8(v8h x, v8h y) {
  return __builtin_shufflevector(x, y, 0,1,2,3,4,5,6,7,8,9,10,11,12,13,14,15);
}
// A-matrix 16x32 f16 per-lane data: halves [hi*8 .. hi*8+7] and [16+hi*8 ..]
__device__ __forceinline__ v16h load_a16(const _Float16* row, int hi) {
  v8h x = *(const v8h*)(row + hi * 8);
  v8h y = *(const v8h*)(row + 16 + hi * 8);
  return cat8(x, y);
}
// B-matrix 32x16 f16 per-lane data: halves [hi*8 .. hi*8+7] and [16+hi*8 ..]
__device__ __forceinline__ v16h load_b16(const _Float16* base, int hi) {
  v8h x = *(const v8h*)(base + hi * 8);
  v8h y = *(const v8h*)(base + 16 + hi * 8);
  return cat8(x, y);
}
#define WMMA(a, b, c) __builtin_amdgcn_wmma_f32_16x16x32_f16(false, a, false, b, (short)0, c, false, false)

// ---------------------------------------------------------------------------
// Kernel 1: kv = x @ kv_w^T   (43904 x 512) x (512 -> 1024)
// 128x64 tile / WG, 8 waves with 2x2 16x16 subtiles each.
// K -> Kws [b][h][n][32]; V -> Vt [b][h][d][352] (transposed for P·V B-frags)
// ---------------------------------------------------------------------------
__global__ void __launch_bounds__(256)
kv_gemm_kernel(const float* __restrict__ x, const float* __restrict__ kv_w,
               _Float16* __restrict__ Kws, _Float16* __restrict__ Vt) {
  __shared__ alignas(16) _Float16 As[128][40];
  __shared__ alignas(16) _Float16 Bs[64][40];
  const int t = threadIdx.x;
  const int lane = t & 31, wid = t >> 5;
  const int lo = lane & 15, hi = lane >> 4;
  const int rowbase = blockIdx.x * 128;  // 343 blocks (exact)
  const int obase   = blockIdx.y * 64;   // 16 blocks over 1024
  const int rtb = (wid >> 1) * 2;        // {0,2,4,6} of 8 row tiles
  const int ctb = (wid & 1) * 2;         // {0,2}     of 4 col tiles
  v8f acc00 = {}, acc01 = {}, acc10 = {}, acc11 = {};
  for (int kb = 0; kb < CDIM; kb += 32) {
    { // stage A: 128 x 32 (f32 -> f16), 16 elements/thread
      int r = t >> 1, c0 = (t & 1) * 16;
      const float* src = x + (size_t)(rowbase + r) * CDIM + kb + c0;
      if (kb + 32 < CDIM) __builtin_prefetch(src + 32, 0, 0);
      #pragma unroll
      for (int i = 0; i < 16; ++i) As[r][c0 + i] = (_Float16)src[i];
    }
    { // stage B: 64 x 32
      int n = t >> 2, c0 = (t & 3) * 8;
      const float* src = kv_w + (size_t)(obase + n) * CDIM + kb + c0;
      #pragma unroll
      for (int i = 0; i < 8; ++i) Bs[n][c0 + i] = (_Float16)src[i];
    }
    __syncthreads();
    v16h a0 = load_a16(&As[(rtb + 0) * 16 + lo][0], hi);
    v16h a1 = load_a16(&As[(rtb + 1) * 16 + lo][0], hi);
    v16h b0 = load_b16(&Bs[(ctb + 0) * 16 + lo][0], hi);
    v16h b1 = load_b16(&Bs[(ctb + 1) * 16 + lo][0], hi);
    acc00 = WMMA(a0, b0, acc00);
    acc01 = WMMA(a0, b1, acc01);
    acc10 = WMMA(a1, b0, acc10);
    acc11 = WMMA(a1, b1, acc11);
    __syncthreads();
  }
  // scatter the four D tiles into K / V-transposed workspaces
  #pragma unroll
  for (int i = 0; i < 4; ++i) {
    v8f acc = (i == 0) ? acc00 : (i == 1) ? acc01 : (i == 2) ? acc10 : acc11;
    int rt = rtb + (i >> 1), ct = ctb + (i & 1);
    #pragma unroll
    for (int v = 0; v < 8; ++v) {
      int R = rowbase + rt * 16 + v + hi * 8;   // < 43904
      int o = obase + ct * 16 + lo;             // < 1024
      int bi = R / NTOK, nn = R - bi * NTOK;
      int h = (o >> 5) & 15, d = o & 31;
      size_t bh = (size_t)bi * NHEADS + h;
      if (o < 512) Kws[(bh * NTOK + nn) * HD + d] = (_Float16)acc[v];
      else         Vt[(bh * HD + d) * VSTR + nn] = (_Float16)acc[v];
    }
  }
}

// ---------------------------------------------------------------------------
// Kernel 2: per (window, head, 32-row block):
//   S = (q*scale)K^T + rel_bias + mask; softmax; O = P V
// ---------------------------------------------------------------------------
__global__ void __launch_bounds__(256)
attn_kernel(const float* __restrict__ qg, const _Float16* __restrict__ Kws,
            const _Float16* __restrict__ Vt, const float* __restrict__ mask,
            const float* __restrict__ rpb, _Float16* __restrict__ Aws) {
  __shared__ alignas(16) _Float16 qs[32][40];
  __shared__ float s[32][NPAD + 2];        // 32 x 354 f32 scoreboard
  __shared__ float obuf[2][2][16][16];     // split-K reduction (ds_add_f32)
  __shared__ float rowinv[32];
  __shared__ int   rterm[32];              // 169*i0 + 13*i1 + i2 per query row
  __shared__ int   cterm[NPAD];            // same per key column

  const int t = threadIdx.x, lane = t & 31, wid = t >> 5;
  const int lo = lane & 15, hi = lane >> 4;
  const int r0 = blockIdx.x * 32;
  const int h  = blockIdx.y;
  const int bz = blockIdx.z;
  const int Bidx = bz >> 6;                // b // nw
  const int w    = bz & 63;                // mask window
  const size_t bh = (size_t)bz * NHEADS + h;
  const float scale = 0.17677669529663687f;  // 32^-0.5

  for (int i = t; i < 2*2*16*16; i += 256) ((float*)obuf)[i] = 0.f;
  for (int i = t; i < NPAD; i += 256) {    // separable rel-pos terms
    int p = (i < NTOK) ? i : 0;
    int j0 = p / 49, jr = p - j0 * 49, j1 = jr / 7, j2 = jr - j1 * 7;
    cterm[i] = 169 * j0 + 13 * j1 + j2;
    if (i < 32) {
      int mg = r0 + i; int q = (mg < NTOK) ? mg : 0;
      int i0 = q / 49, ir = q - i0 * 49, i1 = ir / 7, i2 = ir - i1 * 7;
      rterm[i] = 169 * i0 + 13 * i1 + i2;
    }
  }
  for (int i = t; i < 32 * 32; i += 256) { // stage q (scaled) as f16
    int r = i >> 5, d = i & 31;
    int rg = r0 + r;
    float qv = (rg < NTOK)
        ? qg[(((size_t)Bidx * NHEADS + h) * NTOK + rg) * HD + d] * scale : 0.f;
    qs[r][d] = (_Float16)qv;
  }
  __syncthreads();

  // ---- S = q K^T. One A-frag per wave, streamed over column tiles.
  {
    int rt = wid >> 2;                     // 0..1
    v16h a = load_a16(&qs[rt * 16 + lo][0], hi);
    for (int ct = (wid & 3); ct < 22; ct += 4) {
      int ng = ct * 16 + lo;
      const _Float16* kp = Kws + (bh * NTOK + (ng < NTOK ? ng : 0)) * HD;
      v16h b = load_b16(kp, hi);
      v8f c = {};
      c = WMMA(a, b, c);
      #pragma unroll
      for (int v = 0; v < 8; ++v) {
        int mloc = rt * 16 + v + hi * 8;
        int nloc = ct * 16 + lo;
        int mg = r0 + mloc;
        float add;
        if (nloc >= NTOK)      add = -1e30f;
        else if (mg >= NTOK)   add = 0.f;
        else {
          int idx = rterm[mloc] - cterm[nloc] + 1098;   // 6*(169+13+1)
          add = rpb[idx * NHEADS + h]
              + mask[((size_t)w * NTOK + mg) * NTOK + nloc];
        }
        s[mloc][nloc] = c[v] + add;
      }
    }
  }
  __syncthreads();

  // ---- row softmax: wave owns 4 rows, wave32 shfl reductions
  for (int r = wid * 4; r < wid * 4 + 4; ++r) {
    float m = -1e30f;
    for (int c = lane; c < NPAD; c += 32) m = fmaxf(m, s[r][c]);
    #pragma unroll
    for (int off = 16; off > 0; off >>= 1) m = fmaxf(m, __shfl_xor(m, off, 32));
    float sum = 0.f;
    for (int c = lane; c < NPAD; c += 32) {
      float e = __expf(s[r][c] - m);
      s[r][c] = e;
      sum += e;
    }
    #pragma unroll
    for (int off = 16; off > 0; off >>= 1) sum += __shfl_xor(sum, off, 32);
    if (lane == 0) rowinv[r] = 1.f / sum;
  }
  __syncthreads();

  // ---- O = P V. Wave: rt = wid&1, 4-way split-K (kh = wid>>1),
  //      both hd-tiles from one scaled P A-frag; V B-frags are contiguous
  //      v8h loads thanks to the d-major Vt layout.
  {
    int rt = wid & 1, kh = wid >> 1;       // kh in 0..3
    float rinv = rowinv[rt * 16 + lo];
    const float* srow = &s[rt * 16 + lo][0];
    v8f c0 = {}, c1 = {};
    for (int kt = kh; kt < 11; kt += 4) {
      int kbase = kt * 32;
      v16h a;
      #pragma unroll
      for (int i = 0; i < 16; ++i) {       // a[i] <- k = hi*8 + i + (i&8)
        int k = kbase + hi * 8 + i + (i & 8);
        a[i] = (_Float16)(srow[k] * rinv);
      }
      v16h b0 = load_b16(Vt + (bh * HD +  0 + lo) * VSTR + kbase, hi);
      v16h b1 = load_b16(Vt + (bh * HD + 16 + lo) * VSTR + kbase, hi);
      c0 = WMMA(a, b0, c0);
      c1 = WMMA(a, b1, c1);
    }
    #pragma unroll
    for (int v = 0; v < 8; ++v) {
      atomicAdd(&obuf[rt][0][v + hi * 8][lo], c0[v]);
      atomicAdd(&obuf[rt][1][v + hi * 8][lo], c1[v]);
    }
  }
  __syncthreads();

  // ---- store [b, n, h*32+d] as f16 for projection GEMM
  for (int i = t; i < 1024; i += 256) {
    int rtnt = i >> 8;
    int rt = rtnt >> 1, nt = rtnt & 1;
    int m16 = (i >> 4) & 15, n16 = i & 15;
    int mg = r0 + rt * 16 + m16;
    if (mg < NTOK)
      Aws[((size_t)bz * NTOK + mg) * CDIM + h * HD + nt * 16 + n16] =
          (_Float16)obuf[rt][nt][m16][n16];
  }
}

// ---------------------------------------------------------------------------
// Kernel 3: out = attn_out @ proj_w^T + proj_b   (f32 out, every elem written)
// ---------------------------------------------------------------------------
__global__ void __launch_bounds__(256)
proj_gemm_kernel(const _Float16* __restrict__ Aws, const float* __restrict__ pw,
                 const float* __restrict__ pb, float* __restrict__ out) {
  __shared__ alignas(16) _Float16 As[128][40];
  __shared__ alignas(16) _Float16 Bs[64][40];
  const int t = threadIdx.x, lane = t & 31, wid = t >> 5;
  const int lo = lane & 15, hi = lane >> 4;
  const int rowbase = blockIdx.x * 128;  // 343 blocks
  const int obase   = blockIdx.y * 64;   // 8 blocks over 512
  const int rtb = (wid >> 1) * 2, ctb = (wid & 1) * 2;
  v8f acc00 = {}, acc01 = {}, acc10 = {}, acc11 = {};
  for (int kb = 0; kb < CDIM; kb += 32) {
    { // stage A (already f16): 16 halves/thread as two v8h
      int r = t >> 1, c0 = (t & 1) * 16;
      const _Float16* src = Aws + (size_t)(rowbase + r) * CDIM + kb + c0;
      if (kb + 32 < CDIM) __builtin_prefetch(src + 32, 0, 0);
      *(v8h*)&As[r][c0]     = *(const v8h*)src;
      *(v8h*)&As[r][c0 + 8] = *(const v8h*)(src + 8);
    }
    {
      int n = t >> 2, c0 = (t & 3) * 8;
      const float* src = pw + (size_t)(obase + n) * CDIM + kb + c0;
      #pragma unroll
      for (int i = 0; i < 8; ++i) Bs[n][c0 + i] = (_Float16)src[i];
    }
    __syncthreads();
    v16h a0 = load_a16(&As[(rtb + 0) * 16 + lo][0], hi);
    v16h a1 = load_a16(&As[(rtb + 1) * 16 + lo][0], hi);
    v16h b0 = load_b16(&Bs[(ctb + 0) * 16 + lo][0], hi);
    v16h b1 = load_b16(&Bs[(ctb + 1) * 16 + lo][0], hi);
    acc00 = WMMA(a0, b0, acc00);
    acc01 = WMMA(a0, b1, acc01);
    acc10 = WMMA(a1, b0, acc10);
    acc11 = WMMA(a1, b1, acc11);
    __syncthreads();
  }
  #pragma unroll
  for (int i = 0; i < 4; ++i) {
    v8f acc = (i == 0) ? acc00 : (i == 1) ? acc01 : (i == 2) ? acc10 : acc11;
    int rt = rtb + (i >> 1), ct = ctb + (i & 1);
    #pragma unroll
    for (int v = 0; v < 8; ++v) {
      int R = rowbase + rt * 16 + v + hi * 8;
      int o = obase + ct * 16 + lo;
      out[(size_t)R * CDIM + o] = acc[v] + pb[o];
    }
  }
}

// ---------------------------------------------------------------------------
extern "C" void kernel_launch(void* const* d_in, const int* in_sizes, int n_in,
                              void* d_out, int out_size, void* d_ws, size_t ws_size,
                              hipStream_t stream) {
  // inputs: x, D, H, W, mask, q_global, kv_w, proj_w, proj_b, rpb_table
  const float* x      = (const float*)d_in[0];
  const float* mask   = (const float*)d_in[4];
  const float* qg     = (const float*)d_in[5];
  const float* kv_w   = (const float*)d_in[6];
  const float* proj_w = (const float*)d_in[7];
  const float* proj_b = (const float*)d_in[8];
  const float* rpb    = (const float*)d_in[9];

  const size_t kelems  = (size_t)128 * NHEADS * NTOK * HD;   // 22,478,848
  const size_t vtelems = (size_t)128 * NHEADS * HD * VSTR;   // 23,068,672
  _Float16* Kws = (_Float16*)d_ws;
  _Float16* Vt  = Kws + kelems;
  _Float16* Aws = Vt + vtelems;    // [43904][512] f16
  float* out = (float*)d_out;

  kv_gemm_kernel<<<dim3(343, 16), 256, 0, stream>>>(x, kv_w, Kws, Vt);
  attn_kernel<<<dim3(11, NHEADS, 128), 256, 0, stream>>>(qg, Kws, Vt, mask, rpb, Aws);
  proj_gemm_kernel<<<dim3(343, 8), 256, 0, stream>>>(Aws, proj_w, proj_b, out);
}
